// MHA_22325240005289
// MI455X (gfx1250) — matrix operand order repaired and checked by reading, not compiled, not run on hardware
//
#include <hip/hip_runtime.h>
#include <hip/hip_bf16.h>

// ---------------------------------------------------------------------------
// MHA layer for MI455X (gfx1250, wave32, WMMA bf16, async LDS DMA)
// ---------------------------------------------------------------------------

typedef __bf16 bf16;
typedef __attribute__((ext_vector_type(16))) __bf16 v16bf;
typedef __attribute__((ext_vector_type(8)))  __bf16 v8bf;
typedef __attribute__((ext_vector_type(8)))  float   v8f;
typedef __attribute__((ext_vector_type(4)))  int     v4i;

#define B_    2
#define S_    2048
#define C_    1536
#define HQ_   16
#define HKV_  4
#define DQK_  128
#define DV_   96
#define NQKV_ 2944          // HQ*DQK + HKV*DQK + HKV*DV
#define ROWS_ 4096          // B_*S_
#define EPS_  1e-5f
#define CLIP_ 5.0f

#if __has_builtin(__builtin_amdgcn_global_load_async_to_lds_b128)
#define HAS_ASYNC_LDS 1
#else
#define HAS_ASYNC_LDS 0
#endif

// ---------------------------------------------------------------------------
// helpers
// ---------------------------------------------------------------------------
__device__ __forceinline__ v8f zero8() {
  v8f z;
#pragma unroll
  for (int i = 0; i < 8; i++) z[i] = 0.0f;
  return z;
}

__device__ __forceinline__ v8f wmma_bf16(v16bf a, v16bf b, v8f c) {
  return __builtin_amdgcn_wmma_f32_16x16x32_bf16(
      false, a, false, b, (short)0, c, false, false);
}

// 16-byte global -> LDS copy; async DMA (ASYNCcnt) when available.
// Toolchain declares the builtin as (int4 AS1*, int4 AS3*, imm, imm).
__device__ __forceinline__ void cp_g2l_16(const bf16* g, bf16* l) {
#if HAS_ASYNC_LDS
  __builtin_amdgcn_global_load_async_to_lds_b128(
      (__attribute__((address_space(1))) v4i*)(void*)g,
      (__attribute__((address_space(3))) v4i*)(void*)l, 0, 0);
#else
  *(v8bf*)l = *(const v8bf*)g;
#endif
}

__device__ __forceinline__ void async_wait0() {
#if HAS_ASYNC_LDS
#if __has_builtin(__builtin_amdgcn_s_wait_asynccnt)
  __builtin_amdgcn_s_wait_asynccnt(0);
#else
  asm volatile("s_wait_asynccnt 0x0" ::: "memory");
#endif
#endif
}

// A fragment (16x32 bf16, MxK). LDS: [row][K] row-major, `stride` elems.
__device__ __forceinline__ v16bf load_a_frag(const bf16* base, int stride, int kbase) {
  const int lane = threadIdx.x & 31;
  const int row  = lane & 15;
  const int half = lane >> 4;
  const bf16* p  = base + row * stride + kbase + half * 8;
  v8bf lo = *(const v8bf*)(p);
  v8bf hi = *(const v8bf*)(p + 16);
  v16bf a;
#pragma unroll
  for (int i = 0; i < 8; i++) { a[i] = lo[i]; a[i + 8] = hi[i]; }
  return a;
}

// B fragment (32x16 bf16, KxN). LDS: [N][K] (transposed), `stride` elems.
__device__ __forceinline__ v16bf load_b_frag(const bf16* base, int stride, int kbase) {
  const int lane = threadIdx.x & 31;
  const int col  = lane & 15;
  const int half = lane >> 4;
  const bf16* p  = base + col * stride + kbase + half * 8;
  v8bf lo = *(const v8bf*)(p);
  v8bf hi = *(const v8bf*)(p + 16);
  v16bf b;
#pragma unroll
  for (int i = 0; i < 8; i++) { b[i] = lo[i]; b[i + 8] = hi[i]; }
  return b;
}

// ---------------------------------------------------------------------------
// Kernel 1: pack + TRANSPOSE weights -> bf16 WqkvT[NQKV_][C_], WoT[C_][C_]
// (transposed so both GEMM operands stream as contiguous 16B async copies)
// ---------------------------------------------------------------------------
__global__ __launch_bounds__(256) void mha_prep_weights(
    const float* __restrict__ Wq, const float* __restrict__ Wk,
    const float* __restrict__ Wv, const float* __restrict__ Wo,
    bf16* __restrict__ WqkvT, bf16* __restrict__ WoT) {
  const size_t i0     = (size_t)blockIdx.x * blockDim.x + threadIdx.x;
  const size_t stride = (size_t)gridDim.x * blockDim.x;
  for (size_t t = i0; t < (size_t)NQKV_ * C_; t += stride) {
    int n = (int)(t / C_), k = (int)(t % C_);
    float v;
    if (n < 2048)      v = Wq[(size_t)k * 2048 + n];
    else if (n < 2560) v = Wk[(size_t)k * 512 + (n - 2048)];
    else               v = Wv[(size_t)k * 384 + (n - 2560)];
    WqkvT[t] = (bf16)v;
  }
  for (size_t t = i0; t < (size_t)C_ * C_; t += stride) {
    int n = (int)(t / C_), k = (int)(t % C_);
    WoT[t] = (bf16)Wo[(size_t)k * C_ + n];
  }
}

// ---------------------------------------------------------------------------
// Kernel 2: h = rms_bn(x) -> bf16 [ROWS_][C_]
// ---------------------------------------------------------------------------
__global__ __launch_bounds__(256) void mha_prep_h(
    const float* __restrict__ x, const float* __restrict__ g,
    const float* __restrict__ ms, bf16* __restrict__ hb) {
  const size_t i0     = (size_t)blockIdx.x * blockDim.x + threadIdx.x;
  const size_t stride = (size_t)gridDim.x * blockDim.x;
  for (size_t t = i0; t < (size_t)ROWS_ * C_; t += stride) {
    int c = (int)(t % C_);
    hb[t] = (bf16)(x[t] * rsqrtf(ms[c] + EPS_) * g[c]);
  }
}

// ---------------------------------------------------------------------------
// Shared GEMM core: Cacc = A[M][K] @ BT[N][K]^T, 128x128 tile, 8 waves,
// double-buffered async LDS staging.  Epilogue differs per caller.
// ---------------------------------------------------------------------------
#define GEMM_BODY(K_, A_, BT_, ACC_)                                            \
  const int tid = threadIdx.x;                                                  \
  const int wave = tid >> 5, lane = tid & 31;                                   \
  const int wm = wave >> 2, wn = wave & 3;                                      \
  const int bm = blockIdx.x, bn = blockIdx.y;                                   \
  v8f ACC_[4][2];                                                               \
  _Pragma("unroll") for (int t = 0; t < 4; t++)                                 \
  _Pragma("unroll") for (int u = 0; u < 2; u++) ACC_[t][u] = zero8();           \
  auto load_tiles = [&](int k0, int buf) {                                      \
    bf16* dA = sA + buf * (128 * 40);                                           \
    bf16* dB = sB + buf * (128 * 40);                                           \
    _Pragma("unroll") for (int it = 0; it < 2; it++) {                          \
      int c = tid + it * 256;                                                   \
      int r = c >> 2, cc = (c & 3) * 8;                                         \
      cp_g2l_16(A_ + (size_t)(bm * 128 + r) * K_ + k0 + cc, dA + r * 40 + cc);  \
    }                                                                           \
    _Pragma("unroll") for (int it = 0; it < 2; it++) {                          \
      int c = tid + it * 256;                                                   \
      int r = c >> 2, cc = (c & 3) * 8;                                         \
      cp_g2l_16(BT_ + (size_t)(bn * 128 + r) * K_ + k0 + cc, dB + r * 40 + cc); \
    }                                                                           \
  };                                                                            \
  load_tiles(0, 0);                                                             \
  async_wait0();                                                                \
  __syncthreads();                                                              \
  int cur = 0;                                                                  \
  for (int k0 = 0; k0 < K_; k0 += 32) {                                         \
    if (k0 + 32 < K_) load_tiles(k0 + 32, cur ^ 1);                             \
    const bf16* pA = sA + cur * (128 * 40);                                     \
    const bf16* pB = sB + cur * (128 * 40);                                     \
    v16bf af[4], bfg[2];                                                        \
    _Pragma("unroll") for (int t = 0; t < 4; t++)                               \
      af[t] = load_a_frag(pA + (wm * 64 + t * 16) * 40, 40, 0);                 \
    _Pragma("unroll") for (int u = 0; u < 2; u++)                               \
      bfg[u] = load_b_frag(pB + (wn * 32 + u * 16) * 40, 40, 0);                \
    _Pragma("unroll") for (int t = 0; t < 4; t++)                               \
    _Pragma("unroll") for (int u = 0; u < 2; u++)                               \
      ACC_[t][u] = wmma_bf16(af[t], bfg[u], ACC_[t][u]);                        \
    async_wait0();                                                              \
    __syncthreads();                                                            \
    cur ^= 1;                                                                   \
  }

// ---------------------------------------------------------------------------
// Kernel 3: qkv = h @ Wqkv   (f32 out)
// ---------------------------------------------------------------------------
__global__ __launch_bounds__(256) void mha_gemm_qkv(
    const bf16* __restrict__ A, const bf16* __restrict__ BT,
    float* __restrict__ Cout) {
  __shared__ __attribute__((aligned(16))) bf16 sA[2 * 128 * 40];
  __shared__ __attribute__((aligned(16))) bf16 sB[2 * 128 * 40];
  GEMM_BODY(C_, A, BT, acc)
  const int cn = lane & 15, rofs = (lane >> 4) * 8;
#pragma unroll
  for (int t = 0; t < 4; t++)
#pragma unroll
    for (int u = 0; u < 2; u++)
#pragma unroll
      for (int r = 0; r < 8; r++) {
        int row = bm * 128 + wm * 64 + t * 16 + rofs + r;
        int col = bn * 128 + wn * 32 + u * 16 + cn;
        Cout[(size_t)row * NQKV_ + col] = acc[t][u][r];
      }
}

// ---------------------------------------------------------------------------
// Kernel 6: out = rms_bn2(y @ Wo + bo)  (f32 out)
// ---------------------------------------------------------------------------
__global__ __launch_bounds__(256) void mha_gemm_out(
    const bf16* __restrict__ A, const bf16* __restrict__ BT,
    const float* __restrict__ bo, const float* __restrict__ g2,
    const float* __restrict__ ms2, float* __restrict__ Cout) {
  __shared__ __attribute__((aligned(16))) bf16 sA[2 * 128 * 40];
  __shared__ __attribute__((aligned(16))) bf16 sB[2 * 128 * 40];
  GEMM_BODY(C_, A, BT, acc)
  const int cn = lane & 15, rofs = (lane >> 4) * 8;
#pragma unroll
  for (int t = 0; t < 4; t++)
#pragma unroll
    for (int u = 0; u < 2; u++) {
      int col = bn * 128 + wn * 32 + u * 16 + cn;
      float sc = rsqrtf(ms2[col] + EPS_) * g2[col];
      float bb = bo[col];
#pragma unroll
      for (int r = 0; r < 8; r++) {
        int row = bm * 128 + wm * 64 + t * 16 + rofs + r;
        Cout[(size_t)row * C_ + col] = (acc[t][u][r] + bb) * sc;
      }
    }
}

// ---------------------------------------------------------------------------
// Kernel 4: per-head LayerNorm (+RoPE for q,k).  One wave32 per row.
// ---------------------------------------------------------------------------
__device__ __forceinline__ float wave_sum32(float v) {
#pragma unroll
  for (int m = 1; m < 32; m <<= 1) v += __shfl_xor(v, m, 32);
  return v;
}

__global__ __launch_bounds__(256) void mha_ln_rope(
    const float* __restrict__ qkv,
    const float* __restrict__ qg, const float* __restrict__ qb,
    const float* __restrict__ kg, const float* __restrict__ kb,
    const float* __restrict__ vg, const float* __restrict__ vb,
    bf16* __restrict__ qo, bf16* __restrict__ ko, bf16* __restrict__ vto) {
  const int wid  = blockIdx.x * 8 + (threadIdx.x >> 5);
  const int lane = threadIdx.x & 31;
  const int row  = wid % ROWS_;
  const int unit = wid / ROWS_;     // 0..23: 16 q, 4 k, 4 v
  const int b = row / S_, s = row % S_;
  const float* src = qkv + (size_t)row * NQKV_;
  const float log_step = logf(2048.0f - 64.0f + 1.0f) / 63.0f;

  if (unit < 20) {
    const bool isq = unit < 16;
    const int  h   = isq ? unit : (unit - 16);
    const float* p  = src + (isq ? h * 128 : 2048 + h * 128);
    const float* gw = isq ? qg : kg;
    const float* bw = isq ? qb : kb;
    const int d0 = lane * 4;
    float xv[4];
#pragma unroll
    for (int i = 0; i < 4; i++) xv[i] = p[d0 + i];
    float mu = wave_sum32(xv[0] + xv[1] + xv[2] + xv[3]) * (1.0f / 128.0f);
    float vs = 0.0f;
#pragma unroll
    for (int i = 0; i < 4; i++) { float d = xv[i] - mu; vs += d * d; }
    float inv = rsqrtf(wave_sum32(vs) * (1.0f / 128.0f) + EPS_);
#pragma unroll
    for (int i = 0; i < 4; i++) xv[i] = (xv[i] - mu) * inv * gw[d0 + i] + bw[d0 + i];
    const float pos = (float)s;
#pragma unroll
    for (int pr = 0; pr < 2; pr++) {
      int   f    = (d0 >> 1) + pr;
      float geom = expf((float)f * log_step);
      float th   = pos / ((float)f + geom);
      float cs = cosf(th), sn = sinf(th);
      float e = xv[pr * 2], o = xv[pr * 2 + 1];
      xv[pr * 2]     = e * cs - o * sn;
      xv[pr * 2 + 1] = o * cs + e * sn;
    }
    bf16* dst = (isq ? qo + (((size_t)(b * HQ_ + h)) * S_ + s) * DQK_
                     : ko + (((size_t)(b * HKV_ + h)) * S_ + s) * DQK_) + d0;
#pragma unroll
    for (int i = 0; i < 4; i++) dst[i] = (bf16)xv[i];
  } else {
    const int h = unit - 20;
    const float* p = src + 2560 + h * 96;
    const int d0 = lane * 3;
    float xv[3];
#pragma unroll
    for (int i = 0; i < 3; i++) xv[i] = p[d0 + i];
    float mu = wave_sum32(xv[0] + xv[1] + xv[2]) * (1.0f / 96.0f);
    float vs = 0.0f;
#pragma unroll
    for (int i = 0; i < 3; i++) { float d = xv[i] - mu; vs += d * d; }
    float inv = rsqrtf(wave_sum32(vs) * (1.0f / 96.0f) + EPS_);
#pragma unroll
    for (int i = 0; i < 3; i++) {
      float o = (xv[i] - mu) * inv * vg[d0 + i] + vb[d0 + i];
      vto[((size_t)(b * HKV_ + h) * DV_ + d0 + i) * S_ + s] = (bf16)o;
    }
  }
}

// ---------------------------------------------------------------------------
// Kernel 5: flash attention. 4 waves/block, 64 q rows, kv chunks of 64,
// double-buffered async K/V staging.
// ---------------------------------------------------------------------------
__global__ __launch_bounds__(128) void mha_attn(
    const bf16* __restrict__ qg, const bf16* __restrict__ kg,
    const bf16* __restrict__ vtg, bf16* __restrict__ yout) {
  const int st = blockIdx.x, h = blockIdx.y, b = blockIdx.z;
  const int kvh = h >> 2;
  const bf16* qbase = qg  + (((size_t)(b * HQ_ + h)) * S_ + st * 64) * DQK_;
  const bf16* kbase = kg  + ((size_t)(b * HKV_ + kvh)) * S_ * DQK_;
  const bf16* vbase = vtg + ((size_t)(b * HKV_ + kvh)) * DV_ * S_;

  __shared__ __attribute__((aligned(16))) bf16 sQ[64 * 136];
  __shared__ __attribute__((aligned(16))) bf16 sK[2 * 64 * 136];
  __shared__ __attribute__((aligned(16))) bf16 sVT[2 * 96 * 72];
  __shared__ __attribute__((aligned(16))) bf16 sP[64 * 72];

  const int tid = threadIdx.x, wave = tid >> 5, lane = tid & 31;
  const int m0 = wave * 16;

  auto load_kv = [&](int j0, int buf) {
    bf16* dK = sK + buf * (64 * 136);
    bf16* dV = sVT + buf * (96 * 72);
#pragma unroll
    for (int it = 0; it < 8; it++) {            // K chunk 64x128
      int c = tid + it * 128;
      int r = c >> 4, cc = (c & 15) * 8;
      cp_g2l_16(kbase + (size_t)(j0 + r) * DQK_ + cc, dK + r * 136 + cc);
    }
#pragma unroll
    for (int it = 0; it < 6; it++) {            // V^T chunk 96x64
      int c = tid + it * 128;
      int r = c >> 3, cc = (c & 7) * 8;
      cp_g2l_16(vbase + (size_t)r * S_ + j0 + cc, dV + r * 72 + cc);
    }
  };

  // Q tile 64x128 + first K/V chunk
#pragma unroll
  for (int it = 0; it < 8; it++) {
    int c = tid + it * 128;
    int r = c >> 4, cc = (c & 15) * 8;
    cp_g2l_16(qbase + (size_t)r * DQK_ + cc, sQ + r * 136 + cc);
  }
  load_kv(0, 0);
  async_wait0();
  __syncthreads();

  v8f yacc[6];
#pragma unroll
  for (int d = 0; d < 6; d++) yacc[d] = zero8();
  float mrow[8], lrow[8];
#pragma unroll
  for (int r = 0; r < 8; r++) { mrow[r] = -1e30f; lrow[r] = 0.0f; }
  const float scale = 0.08838834764831845f;  // 1/sqrt(128)

  int cur = 0;
  for (int j0 = 0; j0 < S_; j0 += 64) {
    if (j0 + 64 < S_) load_kv(j0 + 64, cur ^ 1);
    const bf16* pK = sK + cur * (64 * 136);
    const bf16* pV = sVT + cur * (96 * 72);

    // logits: 16 rows x 64 kv per wave
    v8f sacc[4];
#pragma unroll
    for (int n = 0; n < 4; n++) sacc[n] = zero8();
#pragma unroll
    for (int kk = 0; kk < 128; kk += 32) {
      v16bf aq = load_a_frag(sQ + m0 * 136, 136, kk);
#pragma unroll
      for (int n = 0; n < 4; n++) {
        v16bf bk = load_b_frag(pK + n * 16 * 136, 136, kk);
        sacc[n] = wmma_bf16(aq, bk, sacc[n]);
      }
    }
    // scale + tanh clip + online softmax (16-lane half reductions)
    float cmax[8];
#pragma unroll
    for (int r = 0; r < 8; r++) cmax[r] = -1e30f;
#pragma unroll
    for (int n = 0; n < 4; n++)
#pragma unroll
      for (int r = 0; r < 8; r++) {
        float l = sacc[n][r] * scale;
        l = tanhf(l * (1.0f / CLIP_)) * CLIP_;
        sacc[n][r] = l;
        cmax[r] = fmaxf(cmax[r], l);
      }
#pragma unroll
    for (int m = 1; m < 16; m <<= 1)
#pragma unroll
      for (int r = 0; r < 8; r++) cmax[r] = fmaxf(cmax[r], __shfl_xor(cmax[r], m, 32));

    float corr[8], psum[8];
#pragma unroll
    for (int r = 0; r < 8; r++) {
      float mn = fmaxf(mrow[r], cmax[r]);
      corr[r] = expf(mrow[r] - mn);
      mrow[r] = mn;
      psum[r] = 0.0f;
    }
#pragma unroll
    for (int n = 0; n < 4; n++)
#pragma unroll
      for (int r = 0; r < 8; r++) {
        float p = expf(sacc[n][r] - mrow[r]);
        sacc[n][r] = p;
        psum[r] += p;
      }
#pragma unroll
    for (int m = 1; m < 16; m <<= 1)
#pragma unroll
      for (int r = 0; r < 8; r++) psum[r] += __shfl_xor(psum[r], m, 32);
#pragma unroll
    for (int r = 0; r < 8; r++) lrow[r] = lrow[r] * corr[r] + psum[r];
#pragma unroll
    for (int d = 0; d < 6; d++)
#pragma unroll
      for (int r = 0; r < 8; r++) yacc[d][r] *= corr[r];

    // stage P (bf16); rows m0..m0+15 are wave-private, LDS is in-order per
    // wave, so no block barrier needed before the PV matmul.
    {
      const int cn = lane & 15, rofs = (lane >> 4) * 8;
#pragma unroll
      for (int n = 0; n < 4; n++)
#pragma unroll
        for (int r = 0; r < 8; r++)
          sP[(m0 + rofs + r) * 72 + n * 16 + cn] = (bf16)sacc[n][r];
    }

    // y += P @ V
#pragma unroll
    for (int kk = 0; kk < 64; kk += 32) {
      v16bf ap = load_a_frag(sP + m0 * 72, 72, kk);
#pragma unroll
      for (int d = 0; d < 6; d++) {
        v16bf bv = load_b_frag(pV + d * 16 * 72, 72, kk);
        yacc[d] = wmma_bf16(ap, bv, yacc[d]);
      }
    }

    async_wait0();
    __syncthreads();
    cur ^= 1;
  }

  // epilogue: normalize, scatter y[b*S+s][h*96+dv] (bf16)
  const int cn = lane & 15, rofs = (lane >> 4) * 8;
#pragma unroll
  for (int d = 0; d < 6; d++)
#pragma unroll
    for (int r = 0; r < 8; r++) {
      int srow = st * 64 + m0 + rofs + r;
      float yv = yacc[d][r] / lrow[r];
      yout[((size_t)(b * S_ + srow)) * (HQ_ * DV_) + h * DV_ + d * 16 + cn] = (bf16)yv;
    }
}

// ---------------------------------------------------------------------------
// launch
// ---------------------------------------------------------------------------
extern "C" void kernel_launch(void* const* d_in, const int* in_sizes, int n_in,
                              void* d_out, int out_size, void* d_ws, size_t ws_size,
                              hipStream_t stream) {
  const float* x     = (const float*)d_in[0];
  const float* bn1_g = (const float*)d_in[1];
  const float* bn1_ms= (const float*)d_in[2];
  const float* Wq    = (const float*)d_in[3];
  const float* Wk    = (const float*)d_in[4];
  const float* Wv    = (const float*)d_in[5];
  const float* qn_g  = (const float*)d_in[6];
  const float* qn_b  = (const float*)d_in[7];
  const float* kn_g  = (const float*)d_in[8];
  const float* kn_b  = (const float*)d_in[9];
  const float* vn_g  = (const float*)d_in[10];
  const float* vn_b  = (const float*)d_in[11];
  const float* Wo    = (const float*)d_in[12];
  const float* bo    = (const float*)d_in[13];
  const float* bn2_g = (const float*)d_in[14];
  const float* bn2_ms= (const float*)d_in[15];
  float* out = (float*)d_out;

  char* ws = (char*)d_ws;
  size_t off = 0;
  auto alloc = [&](size_t bytes) -> void* {
    void* p = ws + off;
    off += (bytes + 255) & ~(size_t)255;
    return p;
  };
  bf16*  h_bf    = (bf16*) alloc((size_t)ROWS_ * C_ * 2);
  bf16*  WqkvT_bf= (bf16*) alloc((size_t)NQKV_ * C_ * 2);
  bf16*  WoT_bf  = (bf16*) alloc((size_t)C_ * C_ * 2);
  float* qkv_f   = (float*)alloc((size_t)ROWS_ * NQKV_ * 4);
  bf16*  q_bf    = (bf16*) alloc((size_t)B_ * HQ_  * S_ * DQK_ * 2);
  bf16*  k_bf    = (bf16*) alloc((size_t)B_ * HKV_ * S_ * DQK_ * 2);
  bf16*  vT_bf   = (bf16*) alloc((size_t)B_ * HKV_ * DV_ * S_ * 2);
  bf16*  y_bf    = (bf16*) alloc((size_t)ROWS_ * (HQ_ * DV_) * 2);

  mha_prep_weights<<<4096, 256, 0, stream>>>(Wq, Wk, Wv, Wo, WqkvT_bf, WoT_bf);
  mha_prep_h<<<4096, 256, 0, stream>>>(x, bn1_g, bn1_ms, h_bf);
  mha_gemm_qkv<<<dim3(ROWS_ / 128, NQKV_ / 128), 256, 0, stream>>>(h_bf, WqkvT_bf, qkv_f);
  mha_ln_rope<<<(ROWS_ * 24) / 8, 256, 0, stream>>>(qkv_f, qn_g, qn_b, kn_g, kn_b,
                                                    vn_g, vn_b, q_bf, k_bf, vT_bf);
  mha_attn<<<dim3(S_ / 64, HQ_, B_), 128, 0, stream>>>(q_bf, k_bf, vT_bf, y_bf);
  mha_gemm_out<<<dim3(ROWS_ / 128, C_ / 128), 256, 0, stream>>>(y_bf, WoT_bf, bo,
                                                                bn2_g, bn2_ms, out);
}
